// LTMPBlock_87591563035018
// MI455X (gfx1250) — matrix operand order repaired and checked by reading, not compiled, not run on hardware
//
#include <hip/hip_runtime.h>
#include <hip/hip_bf16.h>
#include <math.h>

typedef __attribute__((ext_vector_type(16))) __bf16 v16bf;
typedef __attribute__((ext_vector_type(8)))  __bf16 v8bf;
typedef __attribute__((ext_vector_type(8)))  float  v8f;
typedef __attribute__((ext_vector_type(4)))  unsigned u32x4;
typedef __attribute__((ext_vector_type(8)))  int      i32x8;
typedef __attribute__((ext_vector_type(4)))  int      i32x4;

#define BB 8
#define NT 1024          // tokens
#define CH_DIM 768       // channels
#define HH 12
#define DH 64
#define BN_TOK (BB*NT)   // 8192
#define NHALF 512
#define MERGE_THR 0.9f

static __device__ __forceinline__ v8f wmma_bf16(v16bf a, v16bf b, v8f c) {
  return __builtin_amdgcn_wmma_f32_16x16x32_bf16(false, a, false, b, (short)0, c, false, false);
}
static __device__ __forceinline__ v16bf cat8(v8bf lo, v8bf hi) {
  return __builtin_shufflevector(lo, hi, 0,1,2,3,4,5,6,7,8,9,10,11,12,13,14,15);
}

// TDM: DMA a 2D tile (16-bit elements) from global into LDS (6-arg builtin on this toolchain).
// D# group0: count=1, lds_addr, 57-bit global addr, type=2("image")
// D# group1: data_size=2B, tensor_dim0/1, tile_dim0/1, dim0 stride
static __device__ __forceinline__ void tdm_load_tile_2d(unsigned lds_addr, const void* gaddr,
                                                        int tile_d0, int tile_d1,
                                                        int tensor_d0, int tensor_d1,
                                                        int stride0) {
  unsigned long long ga = (unsigned long long)(size_t)gaddr;
  u32x4 g0;
  g0[0] = 1u;                                             // count=1, user mode
  g0[1] = lds_addr;                                       // LDS byte address
  g0[2] = (unsigned)(ga & 0xFFFFFFFFu);                   // global addr [31:0]
  g0[3] = (unsigned)((ga >> 32) & 0x01FFFFFFu) | (2u << 30);  // addr[56:32] | type=2
  i32x8 g1;
  g1[0] = 1 << 16;                                        // wg_mask=0, data_size=1 (2 bytes)
  g1[1] = (tensor_d0 & 0xFFFF) << 16;                     // tensor_dim0[15:0] @bits[63:48]
  g1[2] = ((tensor_d0 >> 16) & 0xFFFF) | ((tensor_d1 & 0xFFFF) << 16);
  g1[3] = ((tensor_d1 >> 16) & 0xFFFF) | ((tile_d0 & 0xFFFF) << 16);
  g1[4] = (tile_d1 & 0xFFFF);                             // tile_dim1 ; tile_dim2=0
  g1[5] = stride0;                                        // tensor_dim0_stride[31:0]
  g1[6] = 0;
  g1[7] = 0;
  i32x4 z4 = {0, 0, 0, 0};
  i32x8 z8 = {0, 0, 0, 0, 0, 0, 0, 0};
  __builtin_amdgcn_tensor_load_to_lds(g0, g1, z4, z4, z8, 0);
}

// ---------------------------------------------------------------- utility
__global__ __launch_bounds__(256) void zero_f_kernel(float* p, int n) {
  int i = blockIdx.x * 256 + threadIdx.x;
  if (i < n) p[i] = 0.f;
}
__global__ __launch_bounds__(256) void f2bf_kernel(const float* __restrict__ in, __bf16* __restrict__ out, int n) {
  int i = blockIdx.x * 256 + threadIdx.x;
  if (i < n) out[i] = (__bf16)in[i];
}

// ---------------------------------------------------------------- layernorm (row = 768 = 3*256)
__global__ __launch_bounds__(256) void ln_bf16_kernel(const float* __restrict__ x,
                                                      const float* __restrict__ g,
                                                      const float* __restrict__ bta,
                                                      __bf16* __restrict__ out) {
  int row = blockIdx.x;
  int tid = threadIdx.x;
  const float* xr = x + (size_t)row * CH_DIM;
  float v0 = xr[tid], v1 = xr[tid + 256], v2 = xr[tid + 512];
  float s = v0 + v1 + v2;
  float s2 = v0 * v0 + v1 * v1 + v2 * v2;
  for (int m = 16; m >= 1; m >>= 1) { s += __shfl_xor(s, m, 32); s2 += __shfl_xor(s2, m, 32); }
  __shared__ float rs[8], rs2[8];
  if ((tid & 31) == 0) { rs[tid >> 5] = s; rs2[tid >> 5] = s2; }
  __syncthreads();
  float ts = 0.f, ts2 = 0.f;
  #pragma unroll
  for (int i = 0; i < 8; i++) { ts += rs[i]; ts2 += rs2[i]; }
  float mu = ts * (1.f / 768.f);
  float var = ts2 * (1.f / 768.f) - mu * mu;
  float rstd = rsqrtf(var + 1e-5f);
  __bf16* orow = out + (size_t)row * CH_DIM;
  orow[tid]       = (__bf16)((v0 - mu) * rstd * g[tid]       + bta[tid]);
  orow[tid + 256] = (__bf16)((v1 - mu) * rstd * g[tid + 256] + bta[tid + 256]);
  orow[tid + 512] = (__bf16)((v2 - mu) * rstd * g[tid + 512] + bta[tid + 512]);
}

// ---------------------------------------------------------------- generic bf16 WMMA GEMM
// A [M,K] row-major bf16, Bm [K,N] row-major bf16; tiles 128x64x32; 256 thr (8 waves 4x2)
struct GemmP {
  const __bf16* A; const __bf16* Bm; int N; int K;
  const float* bias; const float* resid; const float* size;
  float* outf; __bf16* outb;
  __bf16* q; __bf16* k; __bf16* v;
};

#define EPI_QKV  0
#define EPI_PROJ 1
#define EPI_FC1  2
#define EPI_FC2  3

template <int EPI>
__global__ __launch_bounds__(256) void gemm_bf16_kernel(GemmP p) {
  __shared__ __attribute__((aligned(32))) __bf16 As[128 * 32];
  __shared__ __attribute__((aligned(32))) __bf16 Bt[64 * 32];   // transposed [n][k]
  int tid = threadIdx.x;
  int wave = tid >> 5, lane = tid & 31;
  int lrow = lane & 15, lhalf = lane >> 4;
  int wm = wave >> 1, wn = wave & 1;
  int bm = blockIdx.y * 128, bn = blockIdx.x * 64;
  int K = p.K, N = p.N;
  v8f acc[2][2] = {};
  for (int k0 = 0; k0 < K; k0 += 32) {
    __syncthreads();
    #pragma unroll
    for (int it = 0; it < 2; it++) {
      int idx = tid + it * 256;
      int row = idx >> 2, seg = idx & 3;
      *(v8bf*)&As[row * 32 + seg * 8] =
          *(const v8bf*)&p.A[(size_t)(bm + row) * K + k0 + seg * 8];
    }
    {
      int krow = tid >> 3, nseg = tid & 7;
      v8bf val = *(const v8bf*)&p.Bm[(size_t)(k0 + krow) * N + bn + nseg * 8];
      #pragma unroll
      for (int e = 0; e < 8; e++) Bt[(nseg * 8 + e) * 32 + krow] = val[e];
    }
    if (k0 + 32 < K) {  // prefetch next A tile -> global_prefetch_b8
      int row = tid >> 2, seg = tid & 3;
      __builtin_prefetch(&p.A[(size_t)(bm + row) * K + k0 + 32 + seg * 8], 0, 1);
    }
    __syncthreads();
    #pragma unroll
    for (int i = 0; i < 2; i++) {
      int m0 = wm * 32 + i * 16;
      const __bf16* ar = &As[(m0 + lrow) * 32 + lhalf * 16];
      v16bf af = cat8(*(const v8bf*)ar, *(const v8bf*)(ar + 8));
      #pragma unroll
      for (int j = 0; j < 2; j++) {
        int n0 = wn * 32 + j * 16;
        v16bf bf = *(const v16bf*)&Bt[(n0 + lrow) * 32 + lhalf * 16];
        acc[i][j] = wmma_bf16(af, bf, acc[i][j]);
      }
    }
  }
  // epilogue: lane holds col (lrow), rows lhalf*8 + r
  #pragma unroll
  for (int i = 0; i < 2; i++) {
    #pragma unroll
    for (int j = 0; j < 2; j++) {
      int gc = bn + wn * 32 + j * 16 + lrow;
      #pragma unroll
      for (int r = 0; r < 8; r++) {
        int gr = bm + wm * 32 + i * 16 + lhalf * 8 + r;
        float a = acc[i][j][r];
        if (EPI == EPI_QKV) {
          int which = gc / CH_DIM;
          int cc = gc - which * CH_DIM;
          int h = cc >> 6, d = cc & 63;
          int b = gr >> 10, n = gr & 1023;
          __bf16* dst = (which == 0) ? p.q : ((which == 1) ? p.k : p.v);
          float vv = (which == 0) ? a * 0.125f : a;  // q * Dh^-0.5
          dst[((size_t)(b * HH + h) * NT + n) * DH + d] = (__bf16)vv;
        } else if (EPI == EPI_PROJ) {
          size_t idx = (size_t)gr * CH_DIM + gc;
          p.outf[idx] = (p.resid[idx] + a + p.bias[gc]) * p.size[gr];
        } else if (EPI == EPI_FC1) {
          float t = a + p.bias[gc];
          float ge = 0.5f * t * (1.0f + erff(t * 0.70710678118654752f));
          p.outb[(size_t)gr * 3072 + gc] = (__bf16)ge;
        } else {  // EPI_FC2
          size_t idx = (size_t)gr * CH_DIM + gc;
          p.outf[idx] = p.outf[idx] + a + p.bias[gc];
        }
      }
    }
  }
}

// ---------------------------------------------------------------- flash attention
// grid = B*H*8 ; block 256 (8 waves, one 16-row q tile each => 128 q rows per block)
#define KCH 64
__global__ __launch_bounds__(256) void attn_kernel(const __bf16* __restrict__ qm,
                                                   const __bf16* __restrict__ km,
                                                   const __bf16* __restrict__ vm,
                                                   const float* __restrict__ size_in,
                                                   const float* __restrict__ mask_in,
                                                   float* __restrict__ imp_acc,
                                                   __bf16* __restrict__ xa) {
  __shared__ __attribute__((aligned(32))) __bf16 Kc[KCH * DH];
  __shared__ __attribute__((aligned(32))) __bf16 Vt[DH * KCH];
  __shared__ __attribute__((aligned(32))) __bf16 Pt[8 * 16 * KCH];
  __shared__ float impb[NT];

  int bid = blockIdx.x;
  int qb = bid & 7;
  int bh = bid >> 3;
  int h = bh % HH, b = bh / HH;
  const __bf16* Kp = km + (size_t)bh * NT * DH;
  const __bf16* Vp = vm + (size_t)bh * NT * DH;
  const __bf16* Qp = qm + (size_t)bh * NT * DH;

  int tid = threadIdx.x, wave = tid >> 5, lane = tid & 31;
  int lrow = lane & 15, lhalf = lane >> 4;
  int q0 = qb * 128 + wave * 16;
  unsigned kc_lds = (unsigned)(size_t)&Kc[0];  // flat->LDS: addr[31:0]

  for (int i = tid; i < NT; i += 256) impb[i] = 0.f;

  // Q fragments (Dh=64 -> 2 k-steps of 32)
  v16bf qf[2];
  #pragma unroll
  for (int s = 0; s < 2; s++) {
    const __bf16* base = Qp + (size_t)(q0 + lrow) * DH + s * 32 + lhalf * 16;
    qf[s] = cat8(*(const v8bf*)base, *(const v8bf*)(base + 8));
  }

  v8f mrow, lsum;
  #pragma unroll
  for (int c = 0; c < 8; c++) { mrow[c] = -__builtin_inff(); lsum[c] = 0.f; }

  // -------- pass A: online row max / masked exp-sum (K chunk via TDM DMA)
  for (int ch = 0; ch < NT / KCH; ch++) {
    __syncthreads();
    if (wave == 0) {
      tdm_load_tile_2d(kc_lds, Kp + (size_t)ch * KCH * DH, DH, KCH, DH, NT, DH);
      __builtin_amdgcn_s_wait_tensorcnt(0);
    }
    __syncthreads();
    v8f S[4];
    #pragma unroll
    for (int j = 0; j < 4; j++) {
      v8f a = {};
      #pragma unroll
      for (int s = 0; s < 2; s++) {
        v16bf bf = *(const v16bf*)&Kc[(j * 16 + lrow) * DH + s * 32 + lhalf * 16];
        a = wmma_bf16(qf[s], bf, a);
      }
      int kidx = ch * KCH + j * 16 + lrow;
      float ls = logf(size_in[(b << 10) + kidx]);
      #pragma unroll
      for (int c = 0; c < 8; c++) a[c] += ls;
      S[j] = a;
    }
    v8f mc = S[0];
    #pragma unroll
    for (int j = 1; j < 4; j++)
      #pragma unroll
      for (int c = 0; c < 8; c++) mc[c] = fmaxf(mc[c], S[j][c]);
    for (int m = 1; m < 16; m <<= 1)
      #pragma unroll
      for (int c = 0; c < 8; c++) mc[c] = fmaxf(mc[c], __shfl_xor(mc[c], m, 32));
    v8f mnew, scl;
    #pragma unroll
    for (int c = 0; c < 8; c++) {
      mnew[c] = fmaxf(mrow[c], mc[c]);
      scl[c] = __expf(mrow[c] - mnew[c]);
      lsum[c] *= scl[c];
    }
    v8f es = {};
    #pragma unroll
    for (int j = 0; j < 4; j++) {
      int kidx = ch * KCH + j * 16 + lrow;
      float mk = mask_in[(b << 10) + kidx];
      #pragma unroll
      for (int c = 0; c < 8; c++) es[c] += __expf(S[j][c] - mnew[c]) * mk;
    }
    for (int m = 1; m < 16; m <<= 1)
      #pragma unroll
      for (int c = 0; c < 8; c++) es[c] += __shfl_xor(es[c], m, 32);
    #pragma unroll
    for (int c = 0; c < 8; c++) { lsum[c] += es[c]; mrow[c] = mnew[c]; }
  }

  v8f invl;
  #pragma unroll
  for (int c = 0; c < 8; c++) invl[c] = 1.f / (lsum[c] + 1e-6f);

  // -------- pass B: normalized probs -> imp + PV (TDM for K overlaps manual V transpose)
  v8f O[4] = {};
  for (int ch = 0; ch < NT / KCH; ch++) {
    __syncthreads();
    if (wave == 0)
      tdm_load_tile_2d(kc_lds, Kp + (size_t)ch * KCH * DH, DH, KCH, DH, NT, DH);
    #pragma unroll
    for (int it = 0; it < 2; it++) {
      int idx = tid + it * 256;
      int row = idx >> 3, seg = idx & 7;
      v8bf val = *(const v8bf*)&Vp[(size_t)(ch * KCH + row) * DH + seg * 8];
      #pragma unroll
      for (int e = 0; e < 8; e++) Vt[(seg * 8 + e) * KCH + row] = val[e];
    }
    if (wave == 0) __builtin_amdgcn_s_wait_tensorcnt(0);
    __syncthreads();
    #pragma unroll
    for (int j = 0; j < 4; j++) {
      v8f a = {};
      #pragma unroll
      for (int s = 0; s < 2; s++) {
        v16bf bf = *(const v16bf*)&Kc[(j * 16 + lrow) * DH + s * 32 + lhalf * 16];
        a = wmma_bf16(qf[s], bf, a);
      }
      int kidx = ch * KCH + j * 16 + lrow;
      float ls = logf(size_in[(b << 10) + kidx]);
      float mk = mask_in[(b << 10) + kidx];
      float cs = 0.f;
      #pragma unroll
      for (int c = 0; c < 8; c++) {
        float pv = __expf(a[c] + ls - mrow[c]) * mk * invl[c];
        Pt[wave * 16 * KCH + (lhalf * 8 + c) * KCH + j * 16 + lrow] = (__bf16)pv;
        cs += pv;
      }
      cs += __shfl_xor(cs, 16, 32);
      if (lane < 16) atomicAdd(&impb[kidx], cs);
    }
    #pragma unroll
    for (int t = 0; t < 2; t++) {
      const __bf16* pr = &Pt[wave * 16 * KCH + lrow * KCH + t * 32 + lhalf * 16];
      v16bf af = cat8(*(const v8bf*)pr, *(const v8bf*)(pr + 8));
      #pragma unroll
      for (int j2 = 0; j2 < 4; j2++) {
        v16bf bf = *(const v16bf*)&Vt[(j2 * 16 + lrow) * KCH + t * 32 + lhalf * 16];
        O[j2] = wmma_bf16(af, bf, O[j2]);
      }
    }
  }
  __syncthreads();
  for (int i = tid; i < NT; i += 256) atomicAdd(&imp_acc[(b << 10) + i], impb[i]);
  #pragma unroll
  for (int j2 = 0; j2 < 4; j2++)
    #pragma unroll
    for (int r = 0; r < 8; r++) {
      int qrow = q0 + lhalf * 8 + r;
      xa[((size_t)(b << 10) + qrow) * CH_DIM + h * DH + j2 * 16 + lrow] = (__bf16)O[j2][r];
    }
}

// ---------------------------------------------------------------- metric = normalize(mean_h k) * mask
__global__ __launch_bounds__(256) void metric_kernel(const __bf16* __restrict__ km,
                                                     const float* __restrict__ mask_in,
                                                     float* __restrict__ metric) {
  int gw = blockIdx.x * 8 + (threadIdx.x >> 5);  // token 0..8191
  int lane = threadIdx.x & 31;
  int b = gw >> 10, n = gw & 1023;
  float m0 = 0.f, m1 = 0.f;
  for (int h = 0; h < HH; h++) {
    const __bf16* base = km + ((size_t)(b * HH + h) * NT + n) * DH;
    m0 += (float)base[lane];
    m1 += (float)base[lane + 32];
  }
  m0 *= (1.f / 12.f); m1 *= (1.f / 12.f);
  float ss = m0 * m0 + m1 * m1;
  for (int m = 1; m < 32; m <<= 1) ss += __shfl_xor(ss, m, 32);
  float inv = rsqrtf(ss);
  float mk = mask_in[(b << 10) + n];
  metric[(size_t)gw * DH + lane]      = m0 * inv * mk;
  metric[(size_t)gw * DH + lane + 32] = m1 * inv * mk;
}

// ---------------------------------------------------------------- bipartite matching: one warp per src row
__global__ __launch_bounds__(256) void match_kernel(const float* __restrict__ metric,
                                                    float* __restrict__ merge_mask,
                                                    int* __restrict__ node_idx) {
  int gw = blockIdx.x * 8 + (threadIdx.x >> 5);  // b*512 + i
  int lane = threadIdx.x & 31;
  int b = gw >> 9, i = gw & 511;
  const float* A = metric + ((size_t)(b << 10) + 2 * i) * DH;
  float av0 = A[lane], av1 = A[lane + 32];
  float best = -__builtin_inff();
  int bidx = 0;
  if (i != 0) {
    for (int j = 0; j < NHALF; j++) {
      const float* Bp = metric + ((size_t)(b << 10) + 2 * j + 1) * DH;
      float p = av0 * Bp[lane] + av1 * Bp[lane + 32];
      for (int m = 1; m < 32; m <<= 1) p += __shfl_xor(p, m, 32);
      if (p > best) { best = p; bidx = j; }  // strict > => first index on ties
    }
  }
  if (lane == 0) {
    merge_mask[gw] = (best > MERGE_THR) ? 1.f : 0.f;
    node_idx[gw] = bidx;
  }
}

// ---------------------------------------------------------------- merge stage 1: seed outputs
__global__ __launch_bounds__(256) void c1_seed_kernel(const float* __restrict__ x1,
                                                      const float* __restrict__ size_in,
                                                      const float* __restrict__ mask_in,
                                                      const float* __restrict__ imp_acc,
                                                      const float* __restrict__ merge_mask,
                                                      float* __restrict__ o_x,
                                                      float* __restrict__ o_sz,
                                                      float* __restrict__ o_mk,
                                                      float* __restrict__ imp_src,
                                                      float* __restrict__ imp_dst) {
  int t = blockIdx.x;
  int b = t >> 10, r = t & 1023;
  int orig = (r < NHALF) ? (2 * r) : (2 * (r - NHALF) + 1);
  const float* src = x1 + ((size_t)(b << 10) + orig) * CH_DIM;
  float* dst = o_x + (size_t)t * CH_DIM;
  for (int c = threadIdx.x; c < CH_DIM; c += 256) dst[c] = src[c];
  if (threadIdx.x == 0) {
    o_sz[t] = size_in[(b << 10) + orig];
    float iv = imp_acc[(b << 10) + orig] * (1.f / 12288.f);  // mean over H*N
    if (orig == 0) iv = __builtin_inff();
    if (r < NHALF) {
      imp_src[b * NHALF + r] = iv;
      float mi = mask_in[(b << 10) + orig];
      float mm = merge_mask[b * NHALF + r];
      o_mk[t] = (mi > 0.f) ? (1.f - mm) : mi;
    } else {
      imp_dst[b * NHALF + (r - NHALF)] = iv;
      o_mk[t] = mask_in[(b << 10) + orig];
    }
  }
}

__global__ __launch_bounds__(256) void c1_viz_kernel(const float* __restrict__ viz_in,
                                                     float* __restrict__ o_vz) {
  size_t id = (size_t)blockIdx.x * 256 + threadIdx.x;  // < B*N*N
  int c = (int)(id & 1023);
  int rt = (int)((id >> 10) & 1023);
  int b = (int)(id >> 20);
  int orig = (rt < NHALF) ? (2 * rt) : (2 * (rt - NHALF) + 1);
  o_vz[id] = viz_in[(((size_t)(b << 10) + orig) << 10) + c];
}

// ---------------------------------------------------------------- merge stage 2: scatter (one block per src row)
__global__ __launch_bounds__(256) void c2_scatter_kernel(const float* __restrict__ x1,
                                                         const float* __restrict__ size_in,
                                                         const float* __restrict__ viz_in,
                                                         const float* __restrict__ merge_mask,
                                                         const int* __restrict__ node_idx,
                                                         const float* __restrict__ imp_src,
                                                         float* __restrict__ o_x,
                                                         float* __restrict__ o_sz,
                                                         float* __restrict__ o_vz,
                                                         float* __restrict__ imp_dst) {
  int gb = blockIdx.x;  // b*512 + i
  int b = gb >> 9, i = gb & 511;
  if (merge_mask[gb] == 0.f) return;
  int j = node_idx[gb];
  const float* sx = x1 + ((size_t)(b << 10) + 2 * i) * CH_DIM;
  float* dx = o_x + ((size_t)(b << 10) + NHALF + j) * CH_DIM;
  for (int c = threadIdx.x; c < CH_DIM; c += 256) atomicAdd(&dx[c], sx[c]);
  if (threadIdx.x == 0) {
    atomicAdd(&o_sz[(b << 10) + NHALF + j], size_in[(b << 10) + 2 * i]);
    atomicMax((unsigned int*)&imp_dst[b * NHALF + j], __float_as_uint(imp_src[gb]));
  }
  const float* sv = viz_in + (((size_t)(b << 10) + 2 * i) << 10);
  float* dv = o_vz + (((size_t)(b << 10) + NHALF + j) << 10);
  for (int c = threadIdx.x; c < NT; c += 256)
    atomicMax((unsigned int*)&dv[c], __float_as_uint(sv[c]));
}

// ---------------------------------------------------------------- merge stage 3: x /= size ; prune mask
__global__ __launch_bounds__(256) void c3_final_kernel(float* __restrict__ o_x,
                                                       const float* __restrict__ o_sz,
                                                       float* __restrict__ o_mk,
                                                       const float* __restrict__ imp_src,
                                                       const float* __restrict__ imp_dst) {
  int t = blockIdx.x;
  int b = t >> 10, r = t & 1023;
  float inv = 1.f / o_sz[t];
  float* xr = o_x + (size_t)t * CH_DIM;
  for (int c = threadIdx.x; c < CH_DIM; c += 256) xr[c] *= inv;
  if (threadIdx.x == 0) {
    float ic = (r < NHALF) ? imp_src[b * NHALF + r] : imp_dst[b * NHALF + (r - NHALF)];
    float m = o_mk[t];
    o_mk[t] = (m > 0.f) ? ((ic > 0.f) ? 1.f : 0.f) : m;
  }
}

// ================================================================ host
extern "C" void kernel_launch(void* const* d_in, const int* in_sizes, int n_in,
                              void* d_out, int out_size, void* d_ws, size_t ws_size,
                              hipStream_t stream) {
  const float* x_in    = (const float*)d_in[0];
  const float* size_in = (const float*)d_in[1];
  const float* mask_in = (const float*)d_in[2];
  const float* viz_in  = (const float*)d_in[3];
  const float* qkv_w   = (const float*)d_in[4];
  const float* proj_w  = (const float*)d_in[5];
  const float* proj_b  = (const float*)d_in[6];
  const float* n1g     = (const float*)d_in[7];
  const float* n1b     = (const float*)d_in[8];
  const float* n2g     = (const float*)d_in[9];
  const float* n2b     = (const float*)d_in[10];
  const float* fc1_w   = (const float*)d_in[11];
  const float* fc1_b   = (const float*)d_in[12];
  const float* fc2_w   = (const float*)d_in[13];
  const float* fc2_b   = (const float*)d_in[14];

  float* o_x  = (float*)d_out;                      // B*N*C
  float* o_sz = o_x + (size_t)BN_TOK * CH_DIM;      // B*N
  float* o_mk = o_sz + BN_TOK;                      // B*N
  float* o_vz = o_mk + BN_TOK;                      // B*N*N

  // workspace carve-out
  uint8_t* ws = (uint8_t*)d_ws;
  size_t o = 0;
  auto carve = [&](size_t bytes) { void* p = ws + o; o = (o + bytes + 255) & ~(size_t)255; return p; };
  const size_t TOKC = (size_t)BN_TOK * CH_DIM;  // 6,291,456

  __bf16* wqkv_bf = (__bf16*)carve((size_t)CH_DIM * 3 * CH_DIM * 2);
  __bf16* wproj_bf = (__bf16*)carve((size_t)CH_DIM * CH_DIM * 2);
  __bf16* wfc1_bf = (__bf16*)carve((size_t)CH_DIM * 4 * CH_DIM * 2);
  __bf16* wfc2_bf = (__bf16*)carve((size_t)4 * CH_DIM * CH_DIM * 2);
  __bf16* xn_bf = (__bf16*)carve(TOKC * 2);               // LN1 out, reused for LN2 out
  __bf16* qkva = (__bf16*)carve(4 * TOKC * 2);            // q,k,v,xa ; reused as fc1 act
  __bf16* q_bf = qkva;
  __bf16* k_bf = q_bf + TOKC;
  __bf16* v_bf = k_bf + TOKC;
  __bf16* xa_bf = v_bf + TOKC;
  __bf16* h1_bf = qkva;                                   // alias (exactly 4*TOKC bf16)
  float* x1 = (float*)carve(TOKC * 4);
  float* metric = (float*)carve((size_t)BN_TOK * DH * 4);
  float* imp_acc = (float*)carve((size_t)BN_TOK * 4);
  float* merge_mask = (float*)carve((size_t)BB * NHALF * 4);
  int* node_idx = (int*)carve((size_t)BB * NHALF * 4);
  float* imp_src = (float*)carve((size_t)BB * NHALF * 4);
  float* imp_dst = (float*)carve((size_t)BB * NHALF * 4);
  (void)ws_size; (void)in_sizes; (void)n_in; (void)out_size;

  // 0) zero importance accumulator
  zero_f_kernel<<<(BN_TOK + 255) / 256, 256, 0, stream>>>(imp_acc, BN_TOK);

  // 1) weights -> bf16
  f2bf_kernel<<<(CH_DIM * 3 * CH_DIM + 255) / 256, 256, 0, stream>>>(qkv_w, wqkv_bf, CH_DIM * 3 * CH_DIM);
  f2bf_kernel<<<(CH_DIM * CH_DIM + 255) / 256, 256, 0, stream>>>(proj_w, wproj_bf, CH_DIM * CH_DIM);
  f2bf_kernel<<<(CH_DIM * 4 * CH_DIM + 255) / 256, 256, 0, stream>>>(fc1_w, wfc1_bf, CH_DIM * 4 * CH_DIM);
  f2bf_kernel<<<(4 * CH_DIM * CH_DIM + 255) / 256, 256, 0, stream>>>(fc2_w, wfc2_bf, 4 * CH_DIM * CH_DIM);

  // 2) LN1
  ln_bf16_kernel<<<BN_TOK, 256, 0, stream>>>(x_in, n1g, n1b, xn_bf);

  // 3) QKV GEMM (8192x2304x768) -> q(,scaled),k,v bf16 [B,H,N,Dh]
  {
    GemmP p = {};
    p.A = xn_bf; p.Bm = wqkv_bf; p.N = 3 * CH_DIM; p.K = CH_DIM;
    p.q = q_bf; p.k = k_bf; p.v = v_bf;
    gemm_bf16_kernel<EPI_QKV><<<dim3(36, 64), 256, 0, stream>>>(p);
  }

  // 4) metric = normalize(mean_h k) * mask
  metric_kernel<<<BN_TOK / 8, 256, 0, stream>>>(k_bf, mask_in, metric);

  // 5) attention -> xa bf16, imp_acc
  attn_kernel<<<BB * HH * 8, 256, 0, stream>>>(q_bf, k_bf, v_bf, size_in, mask_in, imp_acc, xa_bf);

  // 6) proj GEMM + residual, x1 = (x + xa@W + b) * size
  {
    GemmP p = {};
    p.A = xa_bf; p.Bm = wproj_bf; p.N = CH_DIM; p.K = CH_DIM;
    p.bias = proj_b; p.resid = x_in; p.size = size_in; p.outf = x1;
    gemm_bf16_kernel<EPI_PROJ><<<dim3(12, 64), 256, 0, stream>>>(p);
  }

  // 7) bipartite matching
  match_kernel<<<BB * NHALF / 8, 256, 0, stream>>>(metric, merge_mask, node_idx);

  // 8) merge: seed, scatter, finalize
  c1_seed_kernel<<<BN_TOK, 256, 0, stream>>>(x1, size_in, mask_in, imp_acc, merge_mask,
                                             o_x, o_sz, o_mk, imp_src, imp_dst);
  c1_viz_kernel<<<(BB * NT * NT) / 256, 256, 0, stream>>>(viz_in, o_vz);
  c2_scatter_kernel<<<BB * NHALF, 256, 0, stream>>>(x1, size_in, viz_in, merge_mask, node_idx,
                                                    imp_src, o_x, o_sz, o_vz, imp_dst);
  c3_final_kernel<<<BN_TOK, 256, 0, stream>>>(o_x, o_sz, o_mk, imp_src, imp_dst);

  // 9) LN2 (reuse xn_bf)
  ln_bf16_kernel<<<BN_TOK, 256, 0, stream>>>(o_x, n2g, n2b, xn_bf);

  // 10) fc1 GEMM (8192x3072x768) + bias + GELU -> h1 bf16
  {
    GemmP p = {};
    p.A = xn_bf; p.Bm = wfc1_bf; p.N = 4 * CH_DIM; p.K = CH_DIM;
    p.bias = fc1_b; p.outb = h1_bf;
    gemm_bf16_kernel<EPI_FC1><<<dim3(48, 64), 256, 0, stream>>>(p);
  }

  // 11) fc2 GEMM (8192x768x3072) + bias, accumulate into o_x
  {
    GemmP p = {};
    p.A = h1_bf; p.Bm = wfc2_bf; p.N = CH_DIM; p.K = 4 * CH_DIM;
    p.bias = fc2_b; p.outf = o_x;
    gemm_bf16_kernel<EPI_FC2><<<dim3(12, 64), 256, 0, stream>>>(p);
  }
}
